// GeneMapper_76106820485632
// MI455X (gfx1250) — matrix operand order repaired and checked by reading, not compiled, not run on hardware
//
#include <hip/hip_runtime.h>
#include <hip/hip_bf16.h>

// ---------------------------------------------------------------------------
// Set-Transformer gene mapper for MI455X (gfx1250, wave32, WMMA).
// All heavy GEMMs use v_wmma_f32_16x16x32_f16 with fp32 accumulation.
// One gene (segment of <=64 reactions x 128 ch) per 256-thread workgroup;
// the whole 4-block pipeline is fused, activations live in LDS (~147 KB,
// dynamic -- WGP has 320 KB).
// Weights are converted once to f16 and packed in WMMA-fragment-major order
// so every B-operand fragment load is 2x global_load_b128, fully coalesced.
// ---------------------------------------------------------------------------

typedef __attribute__((ext_vector_type(16))) _Float16 v16h;
typedef __attribute__((ext_vector_type(8)))  float    v8f;

#define C_    128
#define L_    64
#define H_    4
#define DH_   32
#define G_    4096
#define NB_   4
#define CC_   (C_ * C_)
#define SCALE_ 0.17677669529663687f   // 1/sqrt(32)
#define EPS_   1e-5f

// ---- LDS layout (byte offsets into dynamic smem) --------------------------
#define OFF_XH   0                     // f16 [64][128]  16 KB   activations
#define OFF_QH   16384                 // f16 [64][128]  16 KB
#define OFF_KH   32768                 // f16 [64][128]  16 KB
#define OFF_VH   49152                 // f16 [64][128]  16 KB
#define OFF_PH   65536                 // f16 [4][64][64] 32 KB  (reused as f32 [64][128])
#define OFF_ACC  98304                 // f32 [64][128]  32 KB
#define OFF_SC   131072                // f32 [64][64]   16 KB
#define OFF_TV   147456                // f32 scratch vectors (~2.5 KB)
#define SMEM_BYTES 150016

// ---------------------------------------------------------------------------
// WMMA helpers (wave32; M/N/C layouts per CDNA5 ISA 7.12.2, A/B use contiguous k = 16*(lane/16) + s)
// ---------------------------------------------------------------------------
static __device__ __forceinline__ v8f wmma16(v16h a, v16h b, v8f c) {
  return __builtin_amdgcn_wmma_f32_16x16x32_f16(false, a, false, b, (short)0, c,
                                                false, false);
}

// A fragment: 16x32 f16 tile, row-major with leading dim ld.
// lanes 0-15: M=lane, K in {0..15}; lanes 16-31: M=lane-16, K in {16..31}
static __device__ __forceinline__ v16h load_a(const _Float16* base, int ld) {
  const int lane = threadIdx.x & 31;
  const int hf = lane >> 4, m = lane & 15;
  const _Float16* row = base + m * ld + 16 * hf;
  v16h a;
#pragma unroll
  for (int j = 0; j < 8; ++j) a[j] = row[j];
#pragma unroll
  for (int j = 0; j < 8; ++j) a[8 + j] = row[8 + j];
  return a;
}

// B fragment from packed (fragment-major) weights: each lane's 16 halfs are
// contiguous; consecutive lanes cover consecutive 32B chunks (coalesced).
static __device__ __forceinline__ v16h load_b_packed(const _Float16* blockbase) {
  const int lane = threadIdx.x & 31;
  return *(const v16h*)(blockbase + lane * 16);
}

// B fragment: 32x16 f16 tile from row-major matrix in LDS (leading dim ld).
// lane = column (lane&15); lanes 0-15 hold K=0..15, lanes 16-31 hold K=16..31.
static __device__ __forceinline__ v16h load_b(const _Float16* base, int ld) {
  const int lane = threadIdx.x & 31;
  const int kg = lane >> 4, n = lane & 15;
  const _Float16* p = base + kg * 16 * ld + n;
  v16h b;
#pragma unroll
  for (int j = 0; j < 16; ++j) b[j] = p[j * ld];
  return b;
}

// B = K^T tile: lane reads 16 contiguous halfs of row (n) -> wide ds loads.
// base -> &K[n0][k0], K row-major, 32B-aligned by construction.
static __device__ __forceinline__ v16h load_bT(const _Float16* base, int ld) {
  const int lane = threadIdx.x & 31;
  const int kg = lane >> 4, n = lane & 15;
  return *(const v16h*)(base + n * ld + kg * 16);
}

// [64,128] = A(f16,LDS) @ Wp(f16 packed,global) + bias -> f16 LDS.
static __device__ __forceinline__ void proj_f16(const _Float16* A, const _Float16* Wp,
                                                const float* bias, _Float16* D) {
  const int wave = threadIdx.x >> 5, lane = threadIdx.x & 31;
  const int hf = lane >> 4, n = lane & 15;
  for (int t = wave; t < 32; t += 8) {
    const int mt = t >> 3, nt = t & 7;
    v8f c = {};
#pragma unroll
    for (int k = 0; k < 4; ++k)
      c = wmma16(load_a(A + mt * 16 * C_ + k * 32, C_),
                 load_b_packed(Wp + (k * 8 + nt) * 512), c);
    const float bb = bias[nt * 16 + n];
#pragma unroll
    for (int r = 0; r < 8; ++r)
      D[(mt * 16 + r + 8 * hf) * C_ + nt * 16 + n] = (_Float16)(c[r] + bb);
  }
}

// Same but f32 destination.
static __device__ __forceinline__ void proj_f32(const _Float16* A, const _Float16* Wp,
                                                const float* bias, float* D) {
  const int wave = threadIdx.x >> 5, lane = threadIdx.x & 31;
  const int hf = lane >> 4, n = lane & 15;
  for (int t = wave; t < 32; t += 8) {
    const int mt = t >> 3, nt = t & 7;
    v8f c = {};
#pragma unroll
    for (int k = 0; k < 4; ++k)
      c = wmma16(load_a(A + mt * 16 * C_ + k * 32, C_),
                 load_b_packed(Wp + (k * 8 + nt) * 512), c);
    const float bb = bias[nt * 16 + n];
#pragma unroll
    for (int r = 0; r < 8; ++r)
      D[(mt * 16 + r + 8 * hf) * C_ + nt * 16 + n] = c[r] + bb;
  }
}

// D = addsrc + relu(A @ Wp + bias)   (fused FFN)
static __device__ __forceinline__ void proj_ff(const _Float16* A, const _Float16* Wp,
                                               const float* bias, const float* addsrc,
                                               float* D) {
  const int wave = threadIdx.x >> 5, lane = threadIdx.x & 31;
  const int hf = lane >> 4, n = lane & 15;
  for (int t = wave; t < 32; t += 8) {
    const int mt = t >> 3, nt = t & 7;
    v8f c = {};
#pragma unroll
    for (int k = 0; k < 4; ++k)
      c = wmma16(load_a(A + mt * 16 * C_ + k * 32, C_),
                 load_b_packed(Wp + (k * 8 + nt) * 512), c);
    const float bb = bias[nt * 16 + n];
#pragma unroll
    for (int r = 0; r < 8; ++r) {
      const int off = (mt * 16 + r + 8 * hf) * C_ + nt * 16 + n;
      D[off] = addsrc[off] + fmaxf(c[r] + bb, 0.0f);
    }
  }
}

// Row-wise LayerNorm of (masked attn + residual) over [64][128]:
// o32 gets fp32 result, o16 gets f16 copy.
static __device__ __forceinline__ void ln_rows(const float* attn, const _Float16* resid,
                                               const float* gam, const float* bet,
                                               int valid, float* o32, _Float16* o16) {
  const int tid = threadIdx.x;
  if (tid < L_) {
    const int r = tid;
    float m = 0.0f;
    for (int c = 0; c < C_; ++c) {
      const float a = (r < valid) ? attn[r * C_ + c] : 0.0f;
      const float v = a + (float)resid[r * C_ + c];
      o32[r * C_ + c] = v;
      m += v;
    }
    m *= (1.0f / C_);
    float var = 0.0f;
    for (int c = 0; c < C_; ++c) {
      const float d = o32[r * C_ + c] - m;
      var += d * d;
    }
    const float rs = rsqrtf(var * (1.0f / C_) + EPS_);
    for (int c = 0; c < C_; ++c) {
      const float y = (o32[r * C_ + c] - m) * rs * gam[c] + bet[c];
      o32[r * C_ + c] = y;
      o16[r * C_ + c] = (_Float16)y;
    }
  }
}

// Row-wise LayerNorm of [64][128] f32 -> f16 only.
static __device__ __forceinline__ void ln2_rows(const float* x, const float* gam,
                                                const float* bet, _Float16* o16) {
  const int tid = threadIdx.x;
  if (tid < L_) {
    const int r = tid;
    float m = 0.0f;
    for (int c = 0; c < C_; ++c) m += x[r * C_ + c];
    m *= (1.0f / C_);
    float var = 0.0f;
    for (int c = 0; c < C_; ++c) {
      const float d = x[r * C_ + c] - m;
      var += d * d;
    }
    const float rs = rsqrtf(var * (1.0f / C_) + EPS_);
    for (int c = 0; c < C_; ++c)
      o16[r * C_ + c] = (_Float16)((x[r * C_ + c] - m) * rs * gam[c] + bet[c]);
  }
}

// LayerNorm of a single 128-vector (uniform call, all 256 threads).
static __device__ __forceinline__ void ln_vec(const float* x, const float* gam,
                                              const float* bet, float* y, float* scal) {
  const int tid = threadIdx.x;
  if (tid == 0) {
    float m = 0.0f;
    for (int k = 0; k < C_; ++k) m += x[k];
    m *= (1.0f / C_);
    float v = 0.0f;
    for (int k = 0; k < C_; ++k) { const float d = x[k] - m; v += d * d; }
    scal[0] = m;
    scal[1] = rsqrtf(v * (1.0f / C_) + EPS_);
  }
  __syncthreads();
  if (tid < C_) y[tid] = (x[tid] - scal[0]) * scal[1] * gam[tid] + bet[tid];
  __syncthreads();
}

// ---------------------------------------------------------------------------
// Prep kernels
// ---------------------------------------------------------------------------
// Convert f32 weights -> f16 packed in WMMA-fragment-major order:
// packed[((i blocks)..)] layout: per transformer block i (CC_ elems):
//   32 operand-blocks b = kk*8 + nt (kk = K-step, nt = N-tile), each 512 halfs:
//   lane-major (32 lanes x 16 contiguous halfs), where lane=(kg,n) holds
//   W[i][kk*32 + kg*16 + j][nt*16 + n] for j = 0..15.
__global__ void k_pack_weights(const float* wq, const float* wk, const float* wv,
                               const float* wo, const float* wl,
                               _Float16* q, _Float16* k, _Float16* v,
                               _Float16* o, _Float16* l) {
  const int idx = blockIdx.x * blockDim.x + threadIdx.x;
  if (idx >= NB_ * CC_) return;
  const int i    = idx >> 14;        // / CC_
  const int rem  = idx & (CC_ - 1);
  const int b    = rem >> 9;         // operand-block: kk*8 + nt
  const int lane = (rem >> 4) & 31;
  const int j    = rem & 15;
  const int kk = b >> 3, nt = b & 7;
  const int kg = lane >> 4, n = lane & 15;
  const int src = i * CC_ + (kk * 32 + kg * 16 + j) * C_ + nt * 16 + n;
  q[idx] = (_Float16)wq[src];
  k[idx] = (_Float16)wk[src];
  v[idx] = (_Float16)wv[src];
  o[idx] = (_Float16)wo[src];
  l[idx] = (_Float16)wl[src];
}

// gene_idx is sorted: starts[g] = lower_bound(gene_idx, g)
__global__ void k_starts(const int* gene_idx, int* starts, int E_) {
  const int g = blockIdx.x * blockDim.x + threadIdx.x;
  if (g >= G_) return;
  int lo = 0, hi = E_;
  while (lo < hi) {
    const int mid = (lo + hi) >> 1;
    if (gene_idx[mid] < g) lo = mid + 1; else hi = mid;
  }
  starts[g] = lo;
}

// ---------------------------------------------------------------------------
// Main fused kernel: one gene per workgroup
// ---------------------------------------------------------------------------
__global__ __launch_bounds__(256) void k_main(
    const float* __restrict__ rf, const int* __restrict__ rxn_idx,
    const int* __restrict__ starts,
    const _Float16* __restrict__ WqF, const _Float16* __restrict__ WkF,
    const _Float16* __restrict__ WvF, const _Float16* __restrict__ WoF,
    const _Float16* __restrict__ WlF,
    const float* __restrict__ bq, const float* __restrict__ bk,
    const float* __restrict__ bv, const float* __restrict__ bo,
    const float* __restrict__ blin,
    const float* __restrict__ g1, const float* __restrict__ b1,
    const float* __restrict__ g2, const float* __restrict__ b2,
    const float* __restrict__ Wq32, const float* __restrict__ Wv32,
    const float* __restrict__ Wo32, const float* __restrict__ Wl32,
    const float* __restrict__ seed, float* __restrict__ out, int E_) {
  extern __shared__ char smem[];
  _Float16* xh = (_Float16*)(smem + OFF_XH);
  _Float16* qh = (_Float16*)(smem + OFF_QH);
  _Float16* kh = (_Float16*)(smem + OFF_KH);
  _Float16* vh = (_Float16*)(smem + OFF_VH);
  _Float16* ph = (_Float16*)(smem + OFF_PH);   // f16 probs [4][64][64]
  float*    pf = (float*)(smem + OFF_PH);      // reuse as f32 [64][128]
  float*    acc = (float*)(smem + OFF_ACC);
  float*    sc = (float*)(smem + OFF_SC);
  float*    tv = (float*)(smem + OFF_TV);
  float* q2 = tv;            // [128]
  float* at2 = tv + 128;     // [128]
  float* t1 = tv + 256;      // [128]
  float* t2 = tv + 384;      // [128]
  float* scal = tv + 512;    // reduction scalars

  const int tid = threadIdx.x;
  const int wave = tid >> 5, lane = tid & 31;
  const int hf = lane >> 4, nn = lane & 15;
  const int g = blockIdx.x;

  // keep f16 weights hot (L2 prefetch path)
  if (tid == 0) __builtin_prefetch(WqF, 0, 1);

  const int s0 = starts[g];
  const int s1 = (g + 1 < G_) ? starts[g + 1] : E_;
  int valid = s1 - s0;
  if (valid > L_) valid = L_;

  // ---- gather + densify segment into LDS (f16), zero padding -------------
  for (int idx = tid; idx < L_ * C_; idx += 256) {
    const int r = idx >> 7, c = idx & (C_ - 1);
    float v = 0.0f;
    if (r < valid) {
      const int rx = rxn_idx[s0 + r];
      v = rf[rx * C_ + c];
    }
    xh[idx] = (_Float16)v;
  }
  __syncthreads();

  // ---- two encoder SAB blocks --------------------------------------------
  for (int blk = 0; blk < 2; ++blk) {
    const _Float16* Wq_ = WqF + blk * CC_;
    const _Float16* Wk_ = WkF + blk * CC_;
    const _Float16* Wv_ = WvF + blk * CC_;
    const _Float16* Wo_ = WoF + blk * CC_;
    const _Float16* Wl_ = WlF + blk * CC_;

    proj_f16(xh, Wq_, bq + blk * C_, qh);
    proj_f16(xh, Wk_, bk + blk * C_, kh);
    proj_f16(xh, Wv_, bv + blk * C_, vh);
    __syncthreads();

    // scores + masked softmax, head by head (sc reused)
    for (int h = 0; h < H_; ++h) {
      for (int t = wave; t < 16; t += 8) {
        const int mt = t >> 2, nt = t & 3;
        v8f c = {};
        c = wmma16(load_a(qh + mt * 16 * C_ + h * DH_, C_),
                   load_bT(kh + nt * 16 * C_ + h * DH_, C_), c);
#pragma unroll
        for (int r = 0; r < 8; ++r)
          sc[(mt * 16 + r + 8 * hf) * L_ + nt * 16 + nn] = c[r] * SCALE_;
      }
      __syncthreads();
      if (tid < L_) {
        float mx = -1e30f;
        for (int k = 0; k < valid; ++k) mx = fmaxf(mx, sc[tid * L_ + k]);
        float sum = 0.0f;
        for (int k = 0; k < valid; ++k) {
          const float e = __expf(sc[tid * L_ + k] - mx);
          sc[tid * L_ + k] = e;
          sum += e;
        }
        const float inv = 1.0f / sum;
        for (int k = 0; k < L_; ++k)
          ph[h * L_ * L_ + tid * L_ + k] =
              (_Float16)((k < valid) ? sc[tid * L_ + k] * inv : 0.0f);
      }
      __syncthreads();
    }

    // attn = probs @ V   -> acc (f32 [64][128], heads concatenated)
    for (int t = wave; t < 32; t += 8) {
      const int h = t >> 3, mt = (t >> 1) & 3, nt = t & 1;
      v8f c = {};
#pragma unroll
      for (int k = 0; k < 2; ++k)
        c = wmma16(load_a(ph + h * L_ * L_ + mt * 16 * L_ + k * 32, L_),
                   load_b(vh + (k * 32) * C_ + h * DH_ + nt * 16, C_), c);
#pragma unroll
      for (int r = 0; r < 8; ++r)
        acc[(mt * 16 + r + 8 * hf) * C_ + h * DH_ + nt * 16 + nn] = c[r];
    }
    __syncthreads();

    // acc -> f16 (as A operand for Wo)
    for (int idx = tid; idx < L_ * C_; idx += 256) qh[idx] = (_Float16)acc[idx];
    __syncthreads();

    // O = attn @ Wo + bo -> pf
    proj_f32(qh, Wo_, bo + blk * C_, pf);
    __syncthreads();

    // LN1( mask(O) + residual ) -> acc (f32) + qh (f16)
    ln_rows(pf, xh, g1 + blk * C_, b1 + blk * C_, valid, acc, qh);
    __syncthreads();

    // FFN: pf = acc + relu(qh @ Wlin + blin)
    proj_ff(qh, Wl_, blin + blk * C_, acc, pf);
    __syncthreads();

    // LN2 -> new activations xh (f16)
    ln2_rows(pf, g2 + blk * C_, b2 + blk * C_, xh);
    __syncthreads();
  }

  // ---- PMA block (i = 2): single seed query ------------------------------
  {
    const int i = 2;
    proj_f16(xh, WkF + i * CC_, bk + i * C_, kh);
    proj_f16(xh, WvF + i * CC_, bv + i * C_, vh);
    if (tid < C_) {
      float s = bq[i * C_ + tid];
      for (int k = 0; k < C_; ++k) s += seed[k] * Wq32[i * CC_ + k * C_ + tid];
      q2[tid] = s;
    }
    __syncthreads();
    // per-head scores q . k
    {
      const int h = tid >> 6, j = tid & 63;
      float s = 0.0f;
      for (int d = 0; d < DH_; ++d)
        s += q2[h * DH_ + d] * (float)kh[j * C_ + h * DH_ + d];
      sc[h * L_ + j] = s * SCALE_;
    }
    __syncthreads();
    if (tid < H_) {
      const int h = tid;
      float mx = -1e30f;
      for (int j = 0; j < valid; ++j) mx = fmaxf(mx, sc[h * L_ + j]);
      float sum = 0.0f;
      for (int j = 0; j < valid; ++j) {
        const float e = __expf(sc[h * L_ + j] - mx);
        sc[h * L_ + j] = e;
        sum += e;
      }
      const float inv = 1.0f / sum;
      for (int j = 0; j < L_; ++j)
        sc[h * L_ + j] = (j < valid) ? sc[h * L_ + j] * inv : 0.0f;
    }
    __syncthreads();
    if (tid < C_) {
      const int h = tid >> 5;
      float s = 0.0f;
      for (int j = 0; j < L_; ++j) s += sc[h * L_ + j] * (float)vh[j * C_ + tid];
      at2[tid] = s;
    }
    __syncthreads();
    if (tid < C_) {
      float s = bo[i * C_ + tid];
      for (int k = 0; k < C_; ++k) s += at2[k] * Wo32[i * CC_ + k * C_ + tid];
      t1[tid] = s + seed[tid];   // residual with q_in = seed
    }
    __syncthreads();
    ln_vec(t1, g1 + i * C_, b1 + i * C_, t2, scal);
    if (tid < C_) {
      float s = blin[i * C_ + tid];
      for (int k = 0; k < C_; ++k) s += t2[k] * Wl32[i * CC_ + k * C_ + tid];
      t1[tid] = t2[tid] + fmaxf(s, 0.0f);
    }
    __syncthreads();
    ln_vec(t1, g2 + i * C_, b2 + i * C_, t2, scal);   // t2 = PMA output
  }

  // ---- decoder SAB (i = 3, Lq = Lk = 1: softmax over 1 key == identity) --
  {
    const int i = 3;
    if (tid < C_) {
      float s = bv[i * C_ + tid];
      for (int k = 0; k < C_; ++k) s += t2[k] * Wv32[i * CC_ + k * C_ + tid];
      at2[tid] = s;                   // attn == v
    }
    __syncthreads();
    if (tid < C_) {
      float s = bo[i * C_ + tid];
      for (int k = 0; k < C_; ++k) s += at2[k] * Wo32[i * CC_ + k * C_ + tid];
      t1[tid] = s + t2[tid];
    }
    __syncthreads();
    ln_vec(t1, g1 + i * C_, b1 + i * C_, q2, scal);
    if (tid < C_) {
      float s = blin[i * C_ + tid];
      for (int k = 0; k < C_; ++k) s += q2[k] * Wl32[i * CC_ + k * C_ + tid];
      t1[tid] = q2[tid] + fmaxf(s, 0.0f);
    }
    __syncthreads();
    ln_vec(t1, g2 + i * C_, b2 + i * C_, at2, scal);
    if (tid < C_) {
      float v = at2[tid];
      if (isnan(v)) v = 0.0f;
      if (isinf(v)) v = (v > 0.0f) ? 3.4028234663852886e38f : -3.4028234663852886e38f;
      out[g * C_ + tid] = v;
    }
  }
}

// ---------------------------------------------------------------------------
// Host launcher
// ---------------------------------------------------------------------------
extern "C" void kernel_launch(void* const* d_in, const int* in_sizes, int n_in,
                              void* d_out, int out_size, void* d_ws, size_t ws_size,
                              hipStream_t stream) {
  const float* rf    = (const float*)d_in[0];
  const float* Wq    = (const float*)d_in[1];
  const float* Wk    = (const float*)d_in[2];
  const float* Wv    = (const float*)d_in[3];
  const float* Wo    = (const float*)d_in[4];
  const float* bq    = (const float*)d_in[5];
  const float* bk    = (const float*)d_in[6];
  const float* bv    = (const float*)d_in[7];
  const float* bo    = (const float*)d_in[8];
  const float* Wlin  = (const float*)d_in[9];
  const float* blin  = (const float*)d_in[10];
  const float* g1    = (const float*)d_in[11];
  const float* b1    = (const float*)d_in[12];
  const float* g2    = (const float*)d_in[13];
  const float* b2    = (const float*)d_in[14];
  const float* seed  = (const float*)d_in[15];
  const int*   rxn   = (const int*)d_in[16];
  const int*   gidx  = (const int*)d_in[17];
  const int E = in_sizes[16];

  // workspace layout: 5 packed f16 weight tensors (NB*C*C each) + starts[G]
  _Float16* WqF = (_Float16*)d_ws;
  _Float16* WkF = WqF + NB_ * CC_;
  _Float16* WvF = WkF + NB_ * CC_;
  _Float16* WoF = WvF + NB_ * CC_;
  _Float16* WlF = WoF + NB_ * CC_;
  int* starts = (int*)(WlF + NB_ * CC_);

  {
    const int n = NB_ * CC_;
    k_pack_weights<<<(n + 255) / 256, 256, 0, stream>>>(Wq, Wk, Wv, Wo, Wlin,
                                                        WqF, WkF, WvF, WoF, WlF);
  }
  k_starts<<<(G_ + 255) / 256, 256, 0, stream>>>(gidx, starts, E);

  k_main<<<G_, 256, SMEM_BYTES, stream>>>(
      rf, rxn, starts, WqF, WkF, WvF, WoF, WlF, bq, bk, bv, bo, blin,
      g1, b1, g2, b2, Wq, Wv, Wo, Wlin, seed, (float*)d_out, E);
}
